// MultiheadAttention_49091476194150
// MI455X (gfx1250) — matrix operand order repaired and checked by reading, not compiled, not run on hardware
//
#include <hip/hip_runtime.h>
#include <hip/hip_bf16.h>

// ---------------------------------------------------------------------------
// CDNA5 (gfx1250) bf16 WMMA types & fragment helpers
// ---------------------------------------------------------------------------
typedef __attribute__((ext_vector_type(8)))  __bf16 v8bf;
typedef __attribute__((ext_vector_type(16))) __bf16 v16bf;
typedef __attribute__((ext_vector_type(8)))  float  v8f;

union V16U { v16bf v; v8bf h[2]; __bf16 e[16]; };
union V8U  { v8bf v; __bf16 e[8]; };
union BPack { unsigned u; __bf16 e[2]; };

// LDS byte-offset of a generic pointer that points into __shared__ (DS-style
// address, which is what GLOBAL_LOAD_ASYNC_TO_LDS's VDST operand expects).
__device__ __forceinline__ unsigned lds_off(const void* p) {
  return (unsigned)(unsigned long long)(__attribute__((address_space(3))) const void*)p;
}

// Async global -> LDS copy of 16 bytes (GLOBAL_LOAD_ASYNC_TO_LDS_B128, GV mode,
// tracked by ASYNCcnt).
__device__ __forceinline__ void async_copy_b128(unsigned lds_dst, const void* gsrc) {
  asm volatile("global_load_async_to_lds_b128 %0, %1, off"
               :: "v"(lds_dst), "v"((unsigned long long)gsrc)
               : "memory");
}
__device__ __forceinline__ void async_wait0() {
  asm volatile("s_wait_asynccnt 0x0" ::: "memory");
}

// A-matrix 16x32 bf16 fragment (lane = row, two 16B contiguous k-runs).
// Lanes 0-15 -> M=lane, k-half 0; lanes 16-31 -> M=lane-16, k-half 1.
// VGPR0..3 = K[kh*8 .. kh*8+7], VGPR4..7 = K[16+kh*8 .. 16+kh*8+7].
__device__ __forceinline__ v16bf load_a_frag(const __bf16* base, int ld) {
  const int lane = threadIdx.x & 31;
  const int row  = lane & 15;
  const int kh   = lane >> 4;
  const __bf16* p = base + row * ld + kh * 8;
  V16U u;
  u.h[0] = *(const v8bf*)(p);
  u.h[1] = *(const v8bf*)(p + 16);
  return u.v;
}

// B-matrix 32x16 bf16 fragment from LDS stored transposed [n][k]:
// lanes 0-15 -> N=lane, K=0..7,16..23; lanes 16-31 -> N=lane-16, K=8..15,24..31.
__device__ __forceinline__ v16bf load_b_frag(const __bf16* base, int ld) {
  const int lane = threadIdx.x & 31;
  const int n    = lane & 15;
  const int kh   = lane >> 4;
  const __bf16* p = base + n * ld + kh * 8;
  V16U u;
  u.h[0] = *(const v8bf*)(p);
  u.h[1] = *(const v8bf*)(p + 16);
  return u.v;
}

#define WMMA_BF16(a, b, c) \
  __builtin_amdgcn_wmma_f32_16x16x32_bf16(false, (a), false, (b), (short)0, (c), false, false)

// 16-lane reductions (xor masks < 16 stay within each half of the wave32)
__device__ __forceinline__ float rmax16(float v) {
  v = fmaxf(v, __shfl_xor(v, 1, 32));
  v = fmaxf(v, __shfl_xor(v, 2, 32));
  v = fmaxf(v, __shfl_xor(v, 4, 32));
  v = fmaxf(v, __shfl_xor(v, 8, 32));
  return v;
}
__device__ __forceinline__ float rsum16(float v) {
  v += __shfl_xor(v, 1, 32);
  v += __shfl_xor(v, 2, 32);
  v += __shfl_xor(v, 4, 32);
  v += __shfl_xor(v, 8, 32);
  return v;
}

// ---------------------------------------------------------------------------
// fp32 -> bf16 conversion
// ---------------------------------------------------------------------------
__global__ void cvt_f32_bf16(const float* __restrict__ in, __bf16* __restrict__ out, int n) {
  int i = blockIdx.x * blockDim.x + threadIdx.x;
  if (i < n) out[i] = (__bf16)in[i];
}

// ---------------------------------------------------------------------------
// bf16 GEMM: C[M,N] = A[M,K] @ B[K,N] + bias  (M,N mult of 128; K mult of 32)
// 256 threads = 8 waves; block tile 128x128; each wave computes 32x64.
// A tile staged via GLOBAL_LOAD_ASYNC_TO_LDS_B128; B tile transposed in
// software with packed b32 LDS stores.
// ---------------------------------------------------------------------------
template <bool OUT_F32>
__global__ __launch_bounds__(256)
void gemm_bf16_k(const __bf16* __restrict__ A, const __bf16* __restrict__ B,
                 const float* __restrict__ bias, __bf16* __restrict__ outb,
                 float* __restrict__ outf, int M, int N, int K) {
  constexpr int LDT = 40;                 // 32 + 8 pad (keeps 16B alignment)
  __shared__ __bf16 As[128 * LDT];        // [m][k]
  __shared__ __bf16 Bs[128 * LDT];        // [n][k] (transposed)
  const int tid = threadIdx.x;
  const int wid = tid >> 5;
  const int lane = tid & 31;
  const int wm = wid >> 1;                // 0..3 -> row strip of 32
  const int wn = wid & 1;                 // 0..1 -> col strip of 64
  const int rowBase = blockIdx.y * 128;
  const int colBase = blockIdx.x * 128;

  v8f acc[2][4] = {};

  // Static per-thread tile mappings
  const int arow = tid >> 1, acb = (tid & 1) << 4;         // A: 16 elems/thread
  const int bkk = (tid & 15) << 1, bnb = (tid >> 4) << 3;  // B: 2k x 8n /thread

  for (int k0 = 0; k0 < K; k0 += 32) {
    // A tile 128x32: async global->LDS, no VGPR round trip
    {
      const __bf16* src = A + (size_t)(rowBase + arow) * K + k0 + acb;
      const unsigned dst = lds_off(&As[arow * LDT + acb]);
      async_copy_b128(dst, src);
      async_copy_b128(dst + 16, src + 8);
      if (k0 + 32 < K) __builtin_prefetch(src + 32, 0, 0);
    }
    // B tile 32x128 -> transposed into Bs[n][k]; pack k-pairs into b32 stores
    {
      const __bf16* src0 = B + (size_t)(k0 + bkk) * N + colBase + bnb;
      V8U lo, hi;
      lo.v = *(const v8bf*)(src0);
      hi.v = *(const v8bf*)(src0 + N);
#pragma unroll
      for (int i = 0; i < 8; ++i) {
        BPack p;
        p.e[0] = lo.e[i];
        p.e[1] = hi.e[i];
        *(unsigned*)&Bs[(bnb + i) * LDT + bkk] = p.u;
      }
      if (k0 + 32 < K) __builtin_prefetch(src0 + (size_t)32 * N, 0, 0);
    }
    async_wait0();
    __syncthreads();

    v16bf a0 = load_a_frag(As + (32 * wm) * LDT, LDT);
    v16bf a1 = load_a_frag(As + (32 * wm + 16) * LDT, LDT);
#pragma unroll
    for (int j = 0; j < 4; ++j) {
      v16bf b = load_b_frag(Bs + (64 * wn + 16 * j) * LDT, LDT);
      acc[0][j] = WMMA_BF16(a0, b, acc[0][j]);
      acc[1][j] = WMMA_BF16(a1, b, acc[1][j]);
    }
    __syncthreads();
  }

  // Epilogue: C fragment layout -> lane holds (M = i + 8*(lane>=16), N = lane&15)
  const int colL = lane & 15;
  const int rowHalf = (lane >> 4) << 3;
#pragma unroll
  for (int mi = 0; mi < 2; ++mi) {
#pragma unroll
    for (int j = 0; j < 4; ++j) {
      const int gcol = colBase + 64 * wn + 16 * j + colL;
      const float bv = bias[gcol];
#pragma unroll
      for (int i = 0; i < 8; ++i) {
        const int grow = rowBase + 32 * wm + 16 * mi + i + rowHalf;
        const float v = acc[mi][j][i] + bv;
        if (OUT_F32) outf[(size_t)grow * N + gcol] = v;
        else         outb[(size_t)grow * N + gcol] = (__bf16)v;
      }
    }
  }
}

// ---------------------------------------------------------------------------
// Flash attention: qkv bf16 [4096][3072] laid out per-row as 16 heads x
// (q[64] k[64] v[64]).  Block = 128 threads (4 waves) = 64 queries of one
// (batch, head).  Loops over 16 key tiles of 128.  scale^2 = 1/8 is folded
// into the softmax logits so Q/K tiles can be async-copied verbatim.
// ---------------------------------------------------------------------------
__global__ __launch_bounds__(128)
void flash_attn_k(const __bf16* __restrict__ qkv, __bf16* __restrict__ out) {
  constexpr int LDQ = 72;    // 64 + 8 pad
  constexpr int LDV = 136;   // 128 + 8 pad
  __shared__ __bf16 Qs[64 * LDQ];     // [q][c]
  __shared__ __bf16 Ks[128 * LDQ];    // [key][c]  (== B layout for Q@K^T)
  __shared__ __bf16 Vt[64 * LDV];     // [c][key]  (== B layout for P@V)
  __shared__ __bf16 Ps[64 * LDV];     // [q][key]  (A layout for P@V)

  const int tid = threadIdx.x, lane = tid & 31, w = tid >> 5;
  const int bh = blockIdx.y, b = bh >> 4, h = bh & 15;
  const int qbase = b * 2048 + blockIdx.x * 64;
  const size_t ld = 3072;
  const float s2 = 0.125f;  // (attn_ch^-0.25)^2 = 1/sqrt(64)

  // Q tile 64x64: async global->LDS (covered by first in-loop wait+barrier)
  for (int idx = tid; idx < 64 * 8; idx += 128) {
    const int row = idx >> 3, c8 = (idx & 7) << 3;
    async_copy_b128(lds_off(&Qs[row * LDQ + c8]),
                    qkv + (size_t)(qbase + row) * ld + h * 192 + c8);
  }

  float m_st[8], l_st[8];
#pragma unroll
  for (int i = 0; i < 8; ++i) { m_st[i] = -1e30f; l_st[i] = 0.0f; }
  v8f o[4] = {};

  const int colL = lane & 15;
  const int rowHalf = (lane >> 4) << 3;

  for (int kt = 0; kt < 16; ++kt) {
    const int keybase = b * 2048 + kt * 128;
    // K tile 128x64 -> Ks[key][c]: async global->LDS
    for (int idx = tid; idx < 128 * 8; idx += 128) {
      const int row = idx >> 3, c8 = (idx & 7) << 3;
      async_copy_b128(lds_off(&Ks[row * LDQ + c8]),
                      qkv + (size_t)(keybase + row) * ld + h * 192 + 64 + c8);
    }
    // V tile 128x64 -> transposed Vt[c][key]; pack key-pairs into b32 stores
    for (int idx = tid; idx < 512; idx += 128) {
      const int key = (idx & 63) << 1;
      const int c8  = (idx >> 6) << 3;
      const __bf16* src = qkv + (size_t)(keybase + key) * ld + h * 192 + 128 + c8;
      V8U a, bb;
      a.v  = *(const v8bf*)(src);
      bb.v = *(const v8bf*)(src + ld);
#pragma unroll
      for (int i = 0; i < 8; ++i) {
        BPack p;
        p.e[0] = a.e[i];
        p.e[1] = bb.e[i];
        *(unsigned*)&Vt[(c8 + i) * LDV + key] = p.u;
      }
    }
    async_wait0();
    __syncthreads();

    // S strip (16 rows x 128 keys) per wave: 8 col-tiles x 2 k-steps
    v16bf a0 = load_a_frag(Qs + (16 * w) * LDQ, LDQ);
    v16bf a1 = load_a_frag(Qs + (16 * w) * LDQ + 32, LDQ);
    v8f s[8];
#pragma unroll
    for (int j = 0; j < 8; ++j) {
      v8f z = {};
      v16bf b0 = load_b_frag(Ks + (16 * j) * LDQ, LDQ);
      v16bf b1 = load_b_frag(Ks + (16 * j) * LDQ + 32, LDQ);
      z = WMMA_BF16(a0, b0, z);
      z = WMMA_BF16(a1, b1, z);
      s[j] = z;
    }

    // Online softmax: row i (per vgpr) lives in one 16-lane group.
    // Apply scale^2 to the raw logits here (moved out of the Q load).
#pragma unroll
    for (int i = 0; i < 8; ++i) {
      float mx = -1e30f;
#pragma unroll
      for (int j = 0; j < 8; ++j) {
        const float v = s[j][i] * s2;
        s[j][i] = v;
        mx = fmaxf(mx, v);
      }
      mx = rmax16(mx);
      const float mn = fmaxf(m_st[i], mx);
      const float al = __expf(m_st[i] - mn);
      float sum = 0.0f;
#pragma unroll
      for (int j = 0; j < 8; ++j) {
        const float p = __expf(s[j][i] - mn);
        s[j][i] = p;
        sum += p;
      }
      sum = rsum16(sum);
      l_st[i] = l_st[i] * al + sum;
      m_st[i] = mn;
#pragma unroll
      for (int n = 0; n < 4; ++n) o[n][i] *= al;
      // P -> LDS in A-fragment-friendly [q][key] layout (same-wave rows only)
#pragma unroll
      for (int j = 0; j < 8; ++j)
        Ps[(16 * w + i + rowHalf) * LDV + 16 * j + colL] = (__bf16)s[j][i];
    }

    // O strip += P @ V   (k = 128 keys, 4 k-steps of 32)
#pragma unroll
    for (int kk = 0; kk < 4; ++kk) {
      v16bf a = load_a_frag(Ps + (16 * w) * LDV + 32 * kk, LDV);
#pragma unroll
      for (int n = 0; n < 4; ++n) {
        v16bf bv = load_b_frag(Vt + (16 * n) * LDV + 32 * kk, LDV);
        o[n] = WMMA_BF16(a, bv, o[n]);
      }
    }
    __syncthreads();
  }

  // Normalize by row sum and write bf16 attention output [4096][1024]
#pragma unroll
  for (int n = 0; n < 4; ++n) {
#pragma unroll
    for (int i = 0; i < 8; ++i) {
      const float v = o[n][i] / l_st[i];
      const int grow = qbase + 16 * w + i + rowHalf;
      const int gcol = h * 64 + 16 * n + colL;
      out[(size_t)grow * 1024 + gcol] = (__bf16)v;
    }
  }
}

// ---------------------------------------------------------------------------
// Launch: convert -> QKV GEMM -> flash attention -> proj GEMM
// Workspace layout (bytes):
//   x_bf     @ 0         : 4096*1024*2  =  8388608
//   wqkv_bf  @ 8388608   : 1024*3072*2  =  6291456
//   wproj_bf @ 14680064  : 1024*1024*2  =  2097152
//   qkv_bf   @ 16777216  : 4096*3072*2  = 25165824
//   attn_bf  @ 41943040  : 4096*1024*2  =  8388608   (total 50331648)
// ---------------------------------------------------------------------------
extern "C" void kernel_launch(void* const* d_in, const int* in_sizes, int n_in,
                              void* d_out, int out_size, void* d_ws, size_t ws_size,
                              hipStream_t stream) {
  (void)in_sizes; (void)n_in; (void)out_size; (void)ws_size;
  const float* x      = (const float*)d_in[0];
  const float* W_qkv  = (const float*)d_in[1];
  const float* b_qkv  = (const float*)d_in[2];
  const float* W_proj = (const float*)d_in[3];
  const float* b_proj = (const float*)d_in[4];
  float* out = (float*)d_out;

  char* ws = (char*)d_ws;
  __bf16* x_bf     = (__bf16*)(ws);
  __bf16* wqkv_bf  = (__bf16*)(ws + 8388608);
  __bf16* wproj_bf = (__bf16*)(ws + 14680064);
  __bf16* qkv_bf   = (__bf16*)(ws + 16777216);
  __bf16* attn_bf  = (__bf16*)(ws + 41943040);

  const int M = 4096, K = 1024, N1 = 3072, N2 = 1024;

  cvt_f32_bf16<<<(M * K + 255) / 256, 256, 0, stream>>>(x, x_bf, M * K);
  cvt_f32_bf16<<<(K * N1 + 255) / 256, 256, 0, stream>>>(W_qkv, wqkv_bf, K * N1);
  cvt_f32_bf16<<<(K * N2 + 255) / 256, 256, 0, stream>>>(W_proj, wproj_bf, K * N2);

  gemm_bf16_k<false><<<dim3(N1 / 128, M / 128), 256, 0, stream>>>(
      x_bf, wqkv_bf, b_qkv, qkv_bf, nullptr, M, N1, K);

  flash_attn_k<<<dim3(2048 / 64, 32), 128, 0, stream>>>(qkv_bf, attn_bf);

  gemm_bf16_k<true><<<dim3(N2 / 128, M / 128), 256, 0, stream>>>(
      attn_bf, wproj_bf, b_proj, nullptr, out, M, N2, K);
}
